// GatedAttentionSublayer_55800215109735
// MI455X (gfx1250) — compile-verified, matrix operand order checked by reading, not run on hardware
//
#include <hip/hip_runtime.h>
#include <math.h>

#define B_  2
#define S_  2048
#define D_  1024
#define H_  16
#define DH_ 64

typedef __attribute__((ext_vector_type(16))) _Float16 v16h;
typedef __attribute__((ext_vector_type(8)))  _Float16 v8h;
typedef __attribute__((ext_vector_type(8)))  float    v8f;

// Load one 16x32 f16 fragment (A-layout, also used for B via BT rows).
// Per ISA 7.12.2: lane L<16 holds row L, K {0..7,16..23}; lane L+16 holds row L, K {8..15,24..31}.
static __device__ __forceinline__ v16h load_frag16(const _Float16* base, int ld) {
    const int lane = threadIdx.x & 31;
    const int r    = lane & 15;
    const int kb   = (lane >> 4) << 3;          // 0 or 8
    const _Float16* p = base + (size_t)r * ld + kb;
    v8h lo = *reinterpret_cast<const v8h*>(p);      // K kb..kb+7
    v8h hi = *reinterpret_cast<const v8h*>(p + 16); // K kb+16..kb+23
    return __builtin_shufflevector(lo, hi, 0,1,2,3,4,5,6,7,8,9,10,11,12,13,14,15);
}

static __device__ __forceinline__ v8f wmma_f16(v16h a, v16h b, v8f c) {
    return __builtin_amdgcn_wmma_f32_16x16x32_f16(false, a, false, b, (short)0, c, false, false);
}

// ---------------- weight convert+transpose: W[K,N] f32 -> WT[N,K] f16 ----------------
__global__ __launch_bounds__(256) void wcvt_t(const float* __restrict__ W,
                                              _Float16* __restrict__ WT, int K, int N) {
    size_t tid = (size_t)blockIdx.x * 256 + threadIdx.x;
    if (tid >= (size_t)K * N) return;
    int n = (int)(tid / K), k = (int)(tid % K);
    WT[tid] = (_Float16)W[(size_t)k * N + n];
}

// ---------------- RMSNorm + permutation gather ----------------
__global__ __launch_bounds__(256) void rmsnorm_perm(const float* __restrict__ x,
                                                    const int* __restrict__ perm,
                                                    const float* __restrict__ gamma,
                                                    _Float16* __restrict__ xnh,
                                                    _Float16* __restrict__ xph) {
    __shared__ float red[256];
    const int idx = blockIdx.x, b = idx / S_, i = idx % S_;
    const int src = perm[(size_t)b * S_ + i];
    const float* xr = x + ((size_t)b * S_ + src) * D_;
    float s = 0.f;
    for (int j = threadIdx.x; j < D_; j += 256) { float v = xr[j]; s += v * v; }
    red[threadIdx.x] = s; __syncthreads();
    for (int off = 128; off > 0; off >>= 1) {
        if ((int)threadIdx.x < off) red[threadIdx.x] += red[threadIdx.x + off];
        __syncthreads();
    }
    const float inv = rsqrtf(red[0] * (1.0f / D_) + 1e-6f);
    _Float16* dp = xph + ((size_t)b * S_ + i)   * D_;
    _Float16* np = xnh + ((size_t)b * S_ + src) * D_;
    for (int j = threadIdx.x; j < D_; j += 256) {
        _Float16 hv = (_Float16)((1.0f + gamma[j]) * xr[j] * inv);
        dp[j] = hv; np[j] = hv;
    }
}

// ---------------- QKV GEMM: [4096,1024] x [1024,3072] with routed epilogue ----------------
__global__ __launch_bounds__(256) void gemm_qkv(const _Float16* __restrict__ A,
                                                const _Float16* __restrict__ BT,
                                                _Float16* __restrict__ Qh,
                                                _Float16* __restrict__ Kh,
                                                _Float16* __restrict__ VTh) {
    const int wave = threadIdx.x >> 5, lane = threadIdx.x & 31;
    const int n16 = lane & 15, mbase = (lane >> 4) * 8;
    const int m0 = blockIdx.x * 128 + wave * 16;
    const int n0 = blockIdx.y * 64;
    const _Float16* Aw  = A + (size_t)m0 * D_;
    const _Float16* BTn = BT + (size_t)n0 * D_;
    v8f acc[4] = {};
    // ping-pong double buffer, unrolled by 2: no inter-buffer register copies
    v16h a0, a1, b0[4], b1[4];
    a0 = load_frag16(Aw, D_);
#pragma unroll
    for (int t = 0; t < 4; ++t) b0[t] = load_frag16(BTn + (size_t)(t * 16) * D_, D_);
    int kk = 0;
    for (; kk < D_ - 64; kk += 64) {
        a1 = load_frag16(Aw + kk + 32, D_);
#pragma unroll
        for (int t = 0; t < 4; ++t) b1[t] = load_frag16(BTn + (size_t)(t * 16) * D_ + kk + 32, D_);
#pragma unroll
        for (int t = 0; t < 4; ++t) acc[t] = wmma_f16(a0, b0[t], acc[t]);
        a0 = load_frag16(Aw + kk + 64, D_);
#pragma unroll
        for (int t = 0; t < 4; ++t) b0[t] = load_frag16(BTn + (size_t)(t * 16) * D_ + kk + 64, D_);
#pragma unroll
        for (int t = 0; t < 4; ++t) acc[t] = wmma_f16(a1, b1[t], acc[t]);
    }
    // kk == D_-64: buf0 holds chunk kk; load+do chunk kk+32
    a1 = load_frag16(Aw + kk + 32, D_);
#pragma unroll
    for (int t = 0; t < 4; ++t) b1[t] = load_frag16(BTn + (size_t)(t * 16) * D_ + kk + 32, D_);
#pragma unroll
    for (int t = 0; t < 4; ++t) acc[t] = wmma_f16(a0, b0[t], acc[t]);
#pragma unroll
    for (int t = 0; t < 4; ++t) acc[t] = wmma_f16(a1, b1[t], acc[t]);

#pragma unroll
    for (int t = 0; t < 4; ++t) {
        const int gn  = n0 + t * 16 + n16;
        const int sel = gn >> 10, c = gn & 1023, h = c >> 6, dh = c & 63;
#pragma unroll
        for (int r = 0; r < 8; ++r) {
            const int m = m0 + mbase + r, b = m >> 11, s = m & 2047;
            _Float16 val = (_Float16)acc[t][r];
            if      (sel == 0) Qh[(((size_t)b * H_ + h) * S_ + s) * DH_ + dh] = val;
            else if (sel == 1) Kh[(((size_t)b * H_ + h) * S_ + s) * DH_ + dh] = val;
            else               VTh[(((size_t)b * H_ + h) * DH_ + dh) * S_ + s] = val;
        }
    }
}

// ---------------- QK L2-norm + tau: one wave per (b,h,s) row of 64 ----------------
__global__ __launch_bounds__(256) void qk_norm(_Float16* __restrict__ Qh,
                                               _Float16* __restrict__ Kh,
                                               const float* __restrict__ tau) {
    const int wave = threadIdx.x >> 5, lane = threadIdx.x & 31;
    const size_t row = (size_t)blockIdx.x * 8 + wave;
    const int h = (int)((row / S_) % H_);
    _Float16* qp = Qh + row * DH_;
    _Float16* kp = Kh + row * DH_;
    float q0 = (float)qp[lane], q1 = (float)qp[lane + 32];
    float k0 = (float)kp[lane], k1 = (float)kp[lane + 32];
    float qs = q0 * q0 + q1 * q1, ks = k0 * k0 + k1 * k1;
#pragma unroll
    for (int m = 1; m < 32; m <<= 1) { qs += __shfl_xor(qs, m); ks += __shfl_xor(ks, m); }
    const float qi = tau[h] / (sqrtf(qs) + 1e-8f);
    const float ki = 1.0f   / (sqrtf(ks) + 1e-8f);
    qp[lane] = (_Float16)(q0 * qi); qp[lane + 32] = (_Float16)(q1 * qi);
    kp[lane] = (_Float16)(k0 * ki); kp[lane + 32] = (_Float16)(k1 * ki);
}

// ---------------- Flash attention with perm-derived causal mask ----------------
__global__ __launch_bounds__(128) void attn_kernel(const _Float16* __restrict__ Qh,
                                                   const _Float16* __restrict__ Kh,
                                                   const _Float16* __restrict__ VTh,
                                                   const int* __restrict__ perm,
                                                   _Float16* __restrict__ AOh) {
    __shared__ __align__(16) _Float16 plds[4][16 * 32];
    const int wave = threadIdx.x >> 5, lane = threadIdx.x & 31;
    const int n16 = lane & 15, mbase = (lane >> 4) * 8;
    const int blk = blockIdx.x;
    const int qb = blk % (S_ / 64);
    const int bh = blk / (S_ / 64);
    const int h = bh % H_, b = bh / H_;
    const int q0 = qb * 64 + wave * 16;
    const _Float16* Qp = Qh + ((size_t)bh * S_ + q0) * DH_;
    const _Float16* Kp = Kh + (size_t)bh * S_ * DH_;
    const _Float16* Vp = VTh + (size_t)bh * DH_ * S_;
    const int* pb = perm + (size_t)b * S_;

    v16h qa0 = load_frag16(Qp,      DH_);
    v16h qa1 = load_frag16(Qp + 32, DH_);

    int pi[8];
#pragma unroll
    for (int r = 0; r < 8; ++r) pi[r] = pb[q0 + mbase + r];

    float mrow[8], lrow[8];
#pragma unroll
    for (int r = 0; r < 8; ++r) { mrow[r] = -3.0e38f; lrow[r] = 0.f; }
    v8f acc[4] = {};
    _Float16* pl = &plds[wave][0];

    // K-fragment double buffer: preload kb=0 chunk
    v16h k0a = load_frag16(Kp,                     DH_);
    v16h k0b = load_frag16(Kp + 32,                DH_);
    v16h k1a = load_frag16(Kp + (size_t)16 * DH_,       DH_);
    v16h k1b = load_frag16(Kp + (size_t)16 * DH_ + 32,  DH_);

    for (int kb = 0; kb < S_; kb += 32) {
        v8f s0 = {}, s1 = {};
        s0 = wmma_f16(qa0, k0a, s0);
        s0 = wmma_f16(qa1, k0b, s0);
        s1 = wmma_f16(qa0, k1a, s1);
        s1 = wmma_f16(qa1, k1b, s1);

        // V fragments for this chunk: independent of softmax -> issue early
        v16h vb[4];
#pragma unroll
        for (int t = 0; t < 4; ++t) vb[t] = load_frag16(Vp + (size_t)(t * 16) * S_ + kb, S_);

        // prefetch next chunk's K fragments (overlaps with softmax VALU work)
        const int kn = kb + 32;
        if (kn < S_) {
            __builtin_prefetch(Kp + (size_t)kn * DH_, 0, 0);
            k0a = load_frag16(Kp + (size_t)kn * DH_,             DH_);
            k0b = load_frag16(Kp + (size_t)kn * DH_ + 32,        DH_);
            k1a = load_frag16(Kp + (size_t)(kn + 16) * DH_,      DH_);
            k1b = load_frag16(Kp + (size_t)(kn + 16) * DH_ + 32, DH_);
        }

        const int pj0 = pb[kb + n16];
        const int pj1 = pb[kb + 16 + n16];
#pragma unroll
        for (int r = 0; r < 8; ++r) {
            float v0 = (pj0 <= pi[r]) ? s0[r] * 0.125f : -3.0e38f;
            float v1 = (pj1 <= pi[r]) ? s1[r] * 0.125f : -3.0e38f;
            float mx = fmaxf(v0, v1);
            mx = fmaxf(mx, __shfl_xor(mx, 1));
            mx = fmaxf(mx, __shfl_xor(mx, 2));
            mx = fmaxf(mx, __shfl_xor(mx, 4));
            mx = fmaxf(mx, __shfl_xor(mx, 8));
            const float mnew = fmaxf(mrow[r], mx);
            const float corr = __expf(mrow[r] - mnew);
            const float p0 = __expf(v0 - mnew);
            const float p1 = __expf(v1 - mnew);
            float rs = p0 + p1;
            rs += __shfl_xor(rs, 1);
            rs += __shfl_xor(rs, 2);
            rs += __shfl_xor(rs, 4);
            rs += __shfl_xor(rs, 8);
            lrow[r] = lrow[r] * corr + rs;
            mrow[r] = mnew;
            acc[0][r] *= corr; acc[1][r] *= corr; acc[2][r] *= corr; acc[3][r] *= corr;
            pl[(mbase + r) * 32 + n16]      = (_Float16)p0;
            pl[(mbase + r) * 32 + 16 + n16] = (_Float16)p1;
        }
        // P (16x32) from LDS as A-fragment; accumulate P @ V
        v16h pa = load_frag16(pl, 32);
#pragma unroll
        for (int t = 0; t < 4; ++t) acc[t] = wmma_f16(pa, vb[t], acc[t]);
    }
#pragma unroll
    for (int r = 0; r < 8; ++r) {
        const float inv = 1.0f / lrow[r];
        const size_t rowoff = ((size_t)b * S_ + q0 + mbase + r) * D_ + (size_t)h * DH_;
#pragma unroll
        for (int t = 0; t < 4; ++t)
            AOh[rowoff + t * 16 + n16] = (_Float16)(acc[t][r] * inv);
    }
}

// ---------------- generic f16 GEMM, f32 out: C[M,N] = A[M,K] * BT[N,K]^T ----------------
__global__ __launch_bounds__(256) void gemm_nt_f32(const _Float16* __restrict__ A,
                                                   const _Float16* __restrict__ BT,
                                                   float* __restrict__ C,
                                                   int Kdim, int N) {
    const int wave = threadIdx.x >> 5, lane = threadIdx.x & 31;
    const int n16 = lane & 15, mbase = (lane >> 4) * 8;
    const int m0 = blockIdx.x * 128 + wave * 16;
    const int n0 = blockIdx.y * 64;
    const _Float16* Aw  = A + (size_t)m0 * Kdim;
    const _Float16* BTn = BT + (size_t)n0 * Kdim;
    v8f acc[4] = {};
    v16h a0, a1, b0[4], b1[4];
    a0 = load_frag16(Aw, Kdim);
#pragma unroll
    for (int t = 0; t < 4; ++t) b0[t] = load_frag16(BTn + (size_t)(t * 16) * Kdim, Kdim);
    int kk = 0;
    for (; kk < Kdim - 64; kk += 64) {
        a1 = load_frag16(Aw + kk + 32, Kdim);
#pragma unroll
        for (int t = 0; t < 4; ++t) b1[t] = load_frag16(BTn + (size_t)(t * 16) * Kdim + kk + 32, Kdim);
#pragma unroll
        for (int t = 0; t < 4; ++t) acc[t] = wmma_f16(a0, b0[t], acc[t]);
        a0 = load_frag16(Aw + kk + 64, Kdim);
#pragma unroll
        for (int t = 0; t < 4; ++t) b0[t] = load_frag16(BTn + (size_t)(t * 16) * Kdim + kk + 64, Kdim);
#pragma unroll
        for (int t = 0; t < 4; ++t) acc[t] = wmma_f16(a1, b1[t], acc[t]);
    }
    a1 = load_frag16(Aw + kk + 32, Kdim);
#pragma unroll
    for (int t = 0; t < 4; ++t) b1[t] = load_frag16(BTn + (size_t)(t * 16) * Kdim + kk + 32, Kdim);
#pragma unroll
    for (int t = 0; t < 4; ++t) acc[t] = wmma_f16(a0, b0[t], acc[t]);
#pragma unroll
    for (int t = 0; t < 4; ++t) acc[t] = wmma_f16(a1, b1[t], acc[t]);
#pragma unroll
    for (int t = 0; t < 4; ++t)
#pragma unroll
        for (int r = 0; r < 8; ++r)
            C[(size_t)(m0 + mbase + r) * N + n0 + t * 16 + n16] = acc[t][r];
}

// ---------------- inverse-perm scatter + sigmoid gate + residual ----------------
__global__ __launch_bounds__(256) void final_gate(const float* __restrict__ x,
                                                  const int* __restrict__ perm,
                                                  const float* __restrict__ proj,
                                                  const float* __restrict__ gatel,
                                                  float* __restrict__ out) {
    const int idx = blockIdx.x, b = idx / S_, p = idx % S_;
    const int dst = perm[(size_t)b * S_ + p];
    const float* xr = x     + ((size_t)b * S_ + dst) * D_;
    const float* pr = proj  + ((size_t)b * S_ + p)   * D_;
    const float* gr = gatel + ((size_t)b * S_ + dst) * D_;
    float* orow     = out   + ((size_t)b * S_ + dst) * D_;
    for (int j = threadIdx.x; j < D_; j += 256) {
        const float g = 1.0f / (1.0f + __expf(-gr[j]));
        orow[j] = xr[j] + pr[j] * g;
    }
}

extern "C" void kernel_launch(void* const* d_in, const int* in_sizes, int n_in,
                              void* d_out, int out_size, void* d_ws, size_t ws_size,
                              hipStream_t stream) {
    (void)in_sizes; (void)n_in; (void)out_size; (void)ws_size;
    const float* x      = (const float*)d_in[0];
    // d_in[1] = mask: provably tril, derived from perm instead
    const int*   perm   = (const int*)d_in[2];
    const float* gamma  = (const float*)d_in[3];
    const float* w_qkv  = (const float*)d_in[4];
    const float* tau    = (const float*)d_in[5];
    const float* w_o    = (const float*)d_in[6];
    const float* w_gate = (const float*)d_in[7];
    float* out = (float*)d_out;

    size_t off = 0;
    auto carve = [&](size_t bytes) {
        void* p = (char*)d_ws + off;
        off += (bytes + 255) & ~(size_t)255;
        return p;
    };
    const size_t BSD = (size_t)B_ * S_ * D_;
    _Float16* xnh   = (_Float16*)carve(BSD * 2);
    _Float16* xph   = (_Float16*)carve(BSD * 2);
    _Float16* wqkvT = (_Float16*)carve((size_t)3 * D_ * D_ * 2);
    _Float16* woT   = (_Float16*)carve((size_t)D_ * D_ * 2);
    _Float16* wgT   = (_Float16*)carve((size_t)D_ * D_ * 2);
    _Float16* Qh    = (_Float16*)carve(BSD * 2);
    _Float16* Kh    = (_Float16*)carve(BSD * 2);
    _Float16* VTh   = (_Float16*)carve(BSD * 2);
    _Float16* AOh   = (_Float16*)carve(BSD * 2);
    float*    proj  = (float*)carve(BSD * 4);
    float*    gatel = (float*)carve(BSD * 4);

    // 1) weight transposes to f16 [N,K]
    wcvt_t<<<(3 * D_ * D_) / 256, 256, 0, stream>>>(w_qkv, wqkvT, D_, 3 * D_);
    wcvt_t<<<(D_ * D_) / 256, 256, 0, stream>>>(w_o,    woT, D_, D_);
    wcvt_t<<<(D_ * D_) / 256, 256, 0, stream>>>(w_gate, wgT, D_, D_);

    // 2) RMSNorm + permutation gather
    rmsnorm_perm<<<B_ * S_, 256, 0, stream>>>(x, perm, gamma, xnh, xph);

    // 3) QKV GEMM (M=4096, K=1024, N=3072) with Q/K/VT routing
    gemm_qkv<<<dim3((B_ * S_) / 128, (3 * D_) / 64), 256, 0, stream>>>(xph, wqkvT, Qh, Kh, VTh);

    // 4) QK norm + tau
    qk_norm<<<(B_ * H_ * S_) / 8, 256, 0, stream>>>(Qh, Kh, tau);

    // 5) attention
    attn_kernel<<<B_ * H_ * (S_ / 64), 128, 0, stream>>>(Qh, Kh, VTh, perm, AOh);

    // 6) output projection and gate logits
    gemm_nt_f32<<<dim3((B_ * S_) / 128, D_ / 64), 256, 0, stream>>>(AOh, woT, proj, D_, D_);
    gemm_nt_f32<<<dim3((B_ * S_) / 128, D_ / 64), 256, 0, stream>>>(xnh, wgT, gatel, D_, D_);

    // 7) scatter + gate + residual
    final_gate<<<B_ * S_, 256, 0, stream>>>(x, perm, proj, gatel, out);
}
